// Encoder_55490977464569
// MI455X (gfx1250) — matrix operand order repaired and checked by reading, not compiled, not run on hardware
//
#include <hip/hip_runtime.h>
#include <hip/hip_bf16.h>

typedef _Float16 v16h __attribute__((ext_vector_type(16)));
typedef _Float16 v8h  __attribute__((ext_vector_type(8)));
typedef float    v8f  __attribute__((ext_vector_type(8)));

#define HID    128
#define BATCH  16
#define IN_DIM 32

__device__ __forceinline__ v8f wmma32(v16h a, v16h b, v8f c) {
  // (neg_a, A, neg_b, B, c_mod, C, reuse_a, reuse_b)
  return __builtin_amdgcn_wmma_f32_16x16x32_f16(false, a, false, b, (short)0, c,
                                                false, false);
}

// Fast transcendental path: v_rcp_f32 instead of IEEE div sequences.
__device__ __forceinline__ float fastrcp(float x) { return __builtin_amdgcn_rcpf(x); }
__device__ __forceinline__ float sigf(float x) { return fastrcp(1.f + __expf(-x)); }
__device__ __forceinline__ float tanhfast(float x) {
  float cx = fminf(fmaxf(x, -15.f), 15.f);
  float e  = __expf(2.f * cx);
  return (e - 1.f) * fastrcp(e + 1.f);
}

// A fragment: 16xK f16 row-major (LDS). CDNA5 16-bit A layout (CDNA5 ISA):
// lane L: M=L&15, g=L>>4; elem e<8 -> K = kb + g*8 + e ; e>=8 -> K = kb + 16 + g*8 + (e-8).
// => two contiguous 8-half (16B) runs per lane.
__device__ __forceinline__ v16h load_a(const _Float16* base, int K, int kb, int lane) {
  int M = lane & 15, g = lane >> 4;
  const _Float16* p = base + M * K + kb + g * 8;
  v8h lo = *(const v8h*)(p);
  v8h hi = *(const v8h*)(p + 16);
  v16h a;
#pragma unroll
  for (int i = 0; i < 8; ++i) { a[i] = lo[i]; a[i + 8] = hi[i]; }
  return a;
}

// B fragment for D = A*B with B[k][n] = W[n0+n][k], W row-major [HID][K] f16 (global).
// lane L: n=L&15, g=L>>4; elem e<8 -> K = kb + g*8 + e ; e>=8 -> K = kb + 16 + g*8 + (e-8).
__device__ __forceinline__ v16h load_b(const _Float16* __restrict__ W, int K, int n0,
                                       int kb, int lane) {
  int n = lane & 15, g = lane >> 4;
  const _Float16* p = W + (size_t)(n0 + n) * K + kb + g * 8;
  v8h lo = *(const v8h*)(p);
  v8h hi = *(const v8h*)(p + 16);
  v16h b;
#pragma unroll
  for (int i = 0; i < 8; ++i) { b[i] = lo[i]; b[i + 8] = hi[i]; }
  return b;
}

// ---------------------------------------------------------------------------
// One kernel launch per tree level; stream order provides the level sync.
// Block = one node = 8 wave32; wave w owns output columns [16w, 16w+16).
// ---------------------------------------------------------------------------
template <bool LEAF>
__global__ __launch_bounds__(256) void tree_gru_level(
    const float* __restrict__ x_lvl,     // [count][16][32]
    const float* __restrict__ mask_lvl,  // [count][16]
    const _Float16* __restrict__ Wir, const _Float16* __restrict__ Wiz,
    const _Float16* __restrict__ Win, const _Float16* __restrict__ Whr,
    const _Float16* __restrict__ Whz, const _Float16* __restrict__ Whn,
    const float* __restrict__ bR,   // b_ir + b_hr  [128]
    const float* __restrict__ bZ,   // b_iz + b_hz  [128]
    const float* __restrict__ bN,   // b_in + b_hn  [128]
    const _Float16* __restrict__ h_prev,  // [2*count][16][128] f16
    _Float16* __restrict__ h_cur)         // [count][16][128]  f16
{
  __shared__ __align__(16) _Float16 sX[BATCH * IN_DIM];
  __shared__ __align__(16) _Float16 sHL[BATCH * HID], sHR[BATCH * HID];
  __shared__ __align__(16) _Float16 sCS[BATCH * HID], sS[BATCH * HID];
  __shared__ float fHL[BATCH * HID], fHR[BATCH * HID], fCS[BATCH * HID];

  const int tid  = threadIdx.x;
  const int node = blockIdx.x;
  const int lane = tid & 31;
  const int n0   = (tid >> 5) * 16;

  // Stage x (f32 -> f16 LDS)
  {
    const float* xg = x_lvl + (size_t)node * (BATCH * IN_DIM);
    for (int i = tid; i < BATCH * IN_DIM; i += 256) sX[i] = (_Float16)xg[i];
  }
  if (!LEAF) {
    const _Float16* hlg = h_prev + (size_t)(2 * node) * (BATCH * HID);
    const _Float16* hrg = hlg + BATCH * HID;
    int i0 = tid * 8;  // 256 threads * 8 halves = 2048 = BATCH*HID
    v8h a = *(const v8h*)(hlg + i0);
    v8h b = *(const v8h*)(hrg + i0);
#pragma unroll
    for (int j = 0; j < 8; ++j) {
      int   i  = i0 + j;
      float fa = (float)a[j], fb = (float)b[j], fc = fa + fb;
      sHL[i] = a[j]; sHR[i] = b[j]; sCS[i] = (_Float16)fc;
      fHL[i] = fa;   fHR[i] = fb;   fCS[i] = fc;
    }
  }
  __syncthreads();

  v8f accZ = {}, accN = {}, accRL = {}, accRR = {};
  {
    v16h ax = load_a(sX, IN_DIM, 0, lane);
    accZ = wmma32(ax, load_b(Wiz, IN_DIM, n0, 0, lane), accZ);  // xi_z
    accN = wmma32(ax, load_b(Win, IN_DIM, n0, 0, lane), accN);  // xi_n
    if (!LEAF) {
      v8f z8 = {};
      v8f xr = wmma32(ax, load_b(Wir, IN_DIM, n0, 0, lane), z8);  // xi_r
      accRL = xr; accRR = xr;
    }
  }

  const int nc = n0 + (lane & 15);
  const int g  = lane >> 4;

  if (!LEAF) {
#pragma unroll
    for (int kb = 0; kb < HID; kb += 32) {
      v16h bhr = load_b(Whr, HID, n0, kb, lane);
      accRL = wmma32(load_a(sHL, HID, kb, lane), bhr, accRL);
      accRR = wmma32(load_a(sHR, HID, kb, lane), bhr, accRR);
      accZ  = wmma32(load_a(sCS, HID, kb, lane),
                     load_b(Whz, HID, n0, kb, lane), accZ);
    }
    // r gates -> s (f32 elementwise), publish s as f16 A operand
    float br = bR[nc];
#pragma unroll
    for (int v = 0; v < 8; ++v) {
      int   M  = 8 * g + v;
      float rl = sigf(accRL[v] + br);
      float rr = sigf(accRR[v] + br);
      float s  = rl * fHL[M * HID + nc] + rr * fHR[M * HID + nc];
      sS[M * HID + nc] = (_Float16)s;
    }
    __syncthreads();
#pragma unroll
    for (int kb = 0; kb < HID; kb += 32)
      accN = wmma32(load_a(sS, HID, kb, lane),
                    load_b(Whn, HID, n0, kb, lane), accN);
    __syncthreads();  // all waves done reading sS; safe to reuse as out-stage
  }

  float bz = bZ[nc];
  float bn = bN[nc];
#pragma unroll
  for (int v = 0; v < 8; ++v) {
    int   M  = 8 * g + v;
    float z  = sigf(accZ[v] + bz);
    float nn = tanhfast(accN[v] + bn);
    float cs = LEAF ? 0.f : fCS[M * HID + nc];
    float h  = (1.f - z) * nn + z * cs;
    h *= mask_lvl[node * BATCH + M];
    sS[M * HID + nc] = (_Float16)h;  // stage through LDS for coalesced store
  }
  __syncthreads();
  {
    _Float16* hout = h_cur + (size_t)node * (BATCH * HID);
    int i0 = tid * 8;
    *(v8h*)(hout + i0) = *(const v8h*)(sS + i0);  // one b128 store / thread
  }
}

// f32 -> f16 weight repack + fused-bias precompute.
// halves: [Wir 4096][Wiz 4096][Win 4096][Whr 16384][Whz 16384][Whn 16384]
// floats after 61440 halves: [bR 128][bZ 128][bN 128]
__global__ void cvt_weights(const float* __restrict__ Wir, const float* __restrict__ Wiz,
                            const float* __restrict__ Win, const float* __restrict__ Whr,
                            const float* __restrict__ Whz, const float* __restrict__ Whn,
                            const float* __restrict__ b_ir, const float* __restrict__ b_hr,
                            const float* __restrict__ b_iz, const float* __restrict__ b_hz,
                            const float* __restrict__ b_in, const float* __restrict__ b_hn,
                            _Float16* __restrict__ dst, float* __restrict__ bsum) {
  int i = blockIdx.x * blockDim.x + threadIdx.x;
  if (i < 4096)       dst[i] = (_Float16)Wir[i];
  else if (i < 8192)  dst[i] = (_Float16)Wiz[i - 4096];
  else if (i < 12288) dst[i] = (_Float16)Win[i - 8192];
  else if (i < 28672) dst[i] = (_Float16)Whr[i - 12288];
  else if (i < 45056) dst[i] = (_Float16)Whz[i - 28672];
  else if (i < 61440) dst[i] = (_Float16)Whn[i - 45056];
  else if (i < 61568) bsum[i - 61440]       = b_ir[i - 61440] + b_hr[i - 61440];
  else if (i < 61696) bsum[i - 61568 + 128] = b_iz[i - 61568] + b_hz[i - 61568];
  else if (i < 61824) bsum[i - 61696 + 256] = b_in[i - 61696] + b_hn[i - 61696];
}

// Root head: mu = h @ W_mu.T + b_mu ; logvar = h @ W_lv.T + b_lv  (f32, tiny)
__global__ void head_kernel(const _Float16* __restrict__ h_root,
                            const float* __restrict__ W_mu, const float* __restrict__ b_mu,
                            const float* __restrict__ W_lv, const float* __restrict__ b_lv,
                            float* __restrict__ out) {
  __shared__ float sh[BATCH * HID];
  for (int i = threadIdx.x; i < BATCH * HID; i += 256) sh[i] = (float)h_root[i];
  __syncthreads();
  for (int idx = threadIdx.x; idx < 2 * BATCH * 64; idx += 256) {
    int which = idx >> 10;       // 0 = mu, 1 = logvar
    int rem   = idx & 1023;
    int b     = rem >> 6;
    int o     = rem & 63;
    const float* W = which ? W_lv : W_mu;
    float acc = (which ? b_lv : b_mu)[o];
#pragma unroll 4
    for (int k = 0; k < HID; ++k) acc += sh[b * HID + k] * W[o * HID + k];
    out[idx] = acc;
  }
}

extern "C" void kernel_launch(void* const* d_in, const int* in_sizes, int n_in,
                              void* d_out, int out_size, void* d_ws, size_t ws_size,
                              hipStream_t stream) {
  (void)in_sizes; (void)n_in; (void)out_size; (void)ws_size;
  const float* targets = (const float*)d_in[0];
  const float* mask    = (const float*)d_in[1];
  const float* W_ir = (const float*)d_in[2];  const float* b_ir = (const float*)d_in[3];
  const float* W_hr = (const float*)d_in[4];  const float* b_hr = (const float*)d_in[5];
  const float* W_iz = (const float*)d_in[6];  const float* b_iz = (const float*)d_in[7];
  const float* W_hz = (const float*)d_in[8];  const float* b_hz = (const float*)d_in[9];
  const float* W_in = (const float*)d_in[10]; const float* b_in = (const float*)d_in[11];
  const float* W_hn = (const float*)d_in[12]; const float* b_hn = (const float*)d_in[13];
  const float* W_mu = (const float*)d_in[14]; const float* b_mu = (const float*)d_in[15];
  const float* W_lv = (const float*)d_in[16]; const float* b_lv = (const float*)d_in[17];
  float* out = (float*)d_out;

  char* ws = (char*)d_ws;
  _Float16* w16   = (_Float16*)ws;
  _Float16* Wir16 = w16 + 0;
  _Float16* Wiz16 = w16 + 4096;
  _Float16* Win16 = w16 + 8192;
  _Float16* Whr16 = w16 + 12288;
  _Float16* Whz16 = w16 + 28672;
  _Float16* Whn16 = w16 + 45056;
  float* bsum = (float*)(ws + 122880);   // [bR 128][bZ 128][bN 128]
  float* bR = bsum, *bZ = bsum + 128, *bN = bsum + 256;
  // ping-pong f16 hidden-state buffers (even levels -> A, odd -> B)
  _Float16* bufA = (_Float16*)(ws + 131072);
  _Float16* bufB = (_Float16*)(ws + 131072 + (size_t)16384 * 2048 * 2);

  cvt_weights<<<242, 256, 0, stream>>>(W_ir, W_iz, W_in, W_hr, W_hz, W_hn,
                                       b_ir, b_hr, b_iz, b_hz, b_in, b_hn,
                                       w16, bsum);

  for (int lvl = 14; lvl >= 0; --lvl) {
    int    count = 1 << lvl;
    size_t start = (size_t)count - 1;
    _Float16*       hcur  = (lvl & 1) ? bufB : bufA;
    const _Float16* hprev = (lvl & 1) ? bufA : bufB;
    const float* x_lvl = targets + start * (BATCH * IN_DIM);
    const float* m_lvl = mask + start * BATCH;
    if (lvl == 14)
      tree_gru_level<true><<<count, 256, 0, stream>>>(
          x_lvl, m_lvl, Wir16, Wiz16, Win16, Whr16, Whz16, Whn16,
          bR, bZ, bN, (const _Float16*)nullptr, hcur);
    else
      tree_gru_level<false><<<count, 256, 0, stream>>>(
          x_lvl, m_lvl, Wir16, Wiz16, Win16, Whr16, Whz16, Whn16,
          bR, bZ, bN, hprev, hcur);
  }
  head_kernel<<<1, 256, 0, stream>>>(bufA, W_mu, b_mu, W_lv, b_lv, out);
}
